// MultiLayerLSTM_62878321213460
// MI455X (gfx1250) — matrix operand order repaired and checked by reading, not compiled, not run on hardware
//
#include <hip/hip_runtime.h>

// ---------------------------------------------------------------------------
// MI455X (gfx1250) 2-layer LSTM.
//   T=1024, B=64, C=256, H=512, G=4H=2048
// Plan:
//   1) cast x + 4 weight matrices fp32 -> bf16 (once)
//   2) big parallel WMMA GEMM: xproj = X_bf @ Wih_bf^T + (bih+bhh)   (fp32 out)
//      A-panel (64 x K bf16) staged into LDS once per block via TDM
//      tensor_load_to_lds + s_wait_tensorcnt (fallback: cooperative copy)
//   3) persistent single-WGP recurrence kernel: h/c in LDS, per-step
//      [64x2048] = xproj_t + h @ Whh^T via v_wmma_f32_16x16x32_bf16,
//      sigmoid/tanh gates, write y (bf16 for next layer / fp32 to d_out)
// ---------------------------------------------------------------------------

typedef __bf16 bf16_t;
typedef bf16_t v16bf __attribute__((ext_vector_type(16)));
typedef bf16_t v8bf  __attribute__((ext_vector_type(8)));
typedef float  v8f   __attribute__((ext_vector_type(8)));
typedef unsigned int v4u __attribute__((ext_vector_type(4)));
typedef int v8i __attribute__((ext_vector_type(8)));
typedef int v4i __attribute__((ext_vector_type(4)));

#define TSEQ 1024
#define BB   64
#define CC   256
#define HH   512
#define GG   2048  // 4*H

__device__ __forceinline__ v8f splat8(float f) {
  v8f v;
  v[0]=f; v[1]=f; v[2]=f; v[3]=f; v[4]=f; v[5]=f; v[6]=f; v[7]=f;
  return v;
}

__device__ __forceinline__ v16bf cat16(v8bf lo, v8bf hi) {
  return __builtin_shufflevector(lo, hi, 0,1,2,3,4,5,6,7,8,9,10,11,12,13,14,15);
}

__device__ __forceinline__ float sigmoidf_(float x) {
  return 1.0f / (1.0f + __expf(-x));
}

// ---------------------------------------------------------------------------
// TDM: DMA a 2D bf16 tile (tile_d1 rows x tile_d0 elems, row stride
// stride_d0 elems) from global memory into LDS at byte offset lds_off.
// D# encoding per CDNA5 ISA ch.8 (group0: flags/lds/global/type=2,
// group1: data_size=2B, dims/tile/stride, groups 2-3 zero for 2D).
// Toolchain probe: this lane's builtin takes 6 args
//   (v4u g0, v8i g1, v4i g2, v4i g3, v8i extra, i32 cpol)
// ---------------------------------------------------------------------------
#if __has_builtin(__builtin_amdgcn_tensor_load_to_lds)
#define HAVE_TDM 1
__device__ __forceinline__ void tdm_load_2d_bf16(unsigned lds_off,
                                                 const bf16_t* gptr,
                                                 unsigned tile_d0,
                                                 unsigned tile_d1,
                                                 unsigned stride_d0) {
  unsigned long long ga = (unsigned long long)(uintptr_t)gptr;
  v4u g0;
  g0[0] = 1u;                                   // count=1, user descriptor
  g0[1] = lds_off;                              // LDS byte address
  g0[2] = (unsigned)(ga & 0xFFFFFFFFu);         // global_addr[31:0]
  g0[3] = (unsigned)((ga >> 32) & 0x1FFFFFFu) | (2u << 30);  // [56:32]|type=2
  v8i g1;
  g1[0] = (int)(1u << 16);                      // wg_mask=0, data_size=1 (2B)
  g1[1] = (int)((tile_d0 & 0xFFFFu) << 16);     // tensor_dim0[15:0] (=tile_d0)
  g1[2] = (int)((tile_d0 >> 16) | ((tile_d1 & 0xFFFFu) << 16)); // td0 hi|td1 lo
  g1[3] = (int)((tile_d1 >> 16) | ((tile_d0 & 0xFFFFu) << 16)); // td1 hi|tile_dim0
  g1[4] = (int)(tile_d1 & 0xFFFFu);             // tile_dim1 | tile_dim2=0
  g1[5] = (int)stride_d0;                       // tensor_dim0_stride[31:0]
  g1[6] = 0;                                    // stride0 hi | stride1 lo
  g1[7] = 0;                                    // stride1 hi
  v4i g2 = {0, 0, 0, 0};
  v4i g3 = {0, 0, 0, 0};
  v8i g4 = {0, 0, 0, 0, 0, 0, 0, 0};            // unused trailing group
  __builtin_amdgcn_tensor_load_to_lds(g0, g1, g2, g3, g4, 0);
}
#else
#define HAVE_TDM 0
#endif

// ---------------------------------------------------------------------------
// fp32 -> bf16 cast (grid-stride)
// ---------------------------------------------------------------------------
__global__ void cast_f32_bf16(const float* __restrict__ in,
                              bf16_t* __restrict__ out, int n) {
  for (int i = blockIdx.x * blockDim.x + threadIdx.x; i < n;
       i += gridDim.x * blockDim.x)
    out[i] = (bf16_t)in[i];
}

// ---------------------------------------------------------------------------
// xproj GEMM: out[M=65536, N=2048] = A[M,K](bf16) @ W[N,K](bf16)^T + bias
// Block = 256 threads = 8 waves; all 8 waves share one 64-row A panel
// (same m-group), each wave owns a distinct 32-col slice (ng). The panel is
// DMA'd to LDS once per block via TDM, then A fragments come from LDS
// (ds_load_b128) and B fragments stream from L2.
// WMMA bf16 fragment layouts used here (wave32; same contiguous K order for A and B):
//   A 16x32: lane<16 -> row=lane,  K = kt*32+0..15 (contiguous in LDS row)
//            lane>=16-> row=lane-16, K = kt*32+16..31
//   B 32x16: lane<16 -> col=lane,  K = kt*32+0..15 (contiguous in W row)
//            lane>=16-> col=lane-16, K = kt*32+16..31
//   C/D 16x16 f32: lane l holds col l%16; reg r holds row r + 8*(l/16)
// ---------------------------------------------------------------------------
__global__ __launch_bounds__(256)
void gemm_xproj(const bf16_t* __restrict__ A, const bf16_t* __restrict__ W,
                const float* __restrict__ bih, const float* __restrict__ bhh,
                float* __restrict__ out, int K) {
  extern __shared__ char smem[];
  bf16_t* ldsA = (bf16_t*)smem;  // [64][K]

  const int tid  = threadIdx.x;
  const int wid  = tid >> 5;
  const int lane = tid & 31;
  const int lrow = lane & 15;
  const int half = lane >> 4;

  const int gw    = blockIdx.x * 8 + wid;  // 65536 waves total
  const int mgrp  = gw >> 6;               // 1024 groups of 64 rows
  const int ng    = gw & 63;               // 64 groups of 32 cols
  const int mbase = mgrp * 64;
  const int nbase = ng * 32;

  // ---- stage A panel [mbase .. mbase+64) x K into LDS ----
#if HAVE_TDM
  if (wid == 0) {
    tdm_load_2d_bf16(/*lds_off=*/(unsigned)(uintptr_t)ldsA,
                     A + (size_t)mbase * K,
                     /*tile_d0=*/(unsigned)K, /*tile_d1=*/64u,
                     /*stride_d0=*/(unsigned)K);
    __builtin_amdgcn_s_wait_tensorcnt(0);
  }
#else
  for (int i = tid; i < 64 * (K / 8); i += 256) {
    ((v8bf*)ldsA)[i] = ((const v8bf*)(A + (size_t)mbase * K))[i];
  }
#endif

  // bias preload overlaps the panel DMA
  v8f acc[4][2];
  {
    int n0 = nbase + lrow;
    int n1 = nbase + 16 + lrow;
    float bs0 = bih[n0] + bhh[n0];
    float bs1 = bih[n1] + bhh[n1];
    for (int mt = 0; mt < 4; ++mt) {
      acc[mt][0] = splat8(bs0);
      acc[mt][1] = splat8(bs1);
    }
  }

  __syncthreads();  // A panel visible to all 8 waves

  const int kts = K >> 5;
#pragma unroll 4
  for (int kt = 0; kt < kts; ++kt) {
    v16bf a[4], b[2];
    const int ka = kt * 32 + half * 16;
    for (int mt = 0; mt < 4; ++mt) {
      const bf16_t* ap = ldsA + (mt * 16 + lrow) * K + ka;
      v8bf lo = *(const v8bf*)(ap);
      v8bf hi = *(const v8bf*)(ap + 8);
      a[mt] = cat16(lo, hi);
    }
    const int kb = kt * 32 + half * 16;
    for (int s = 0; s < 2; ++s) {
      size_t base = (size_t)(nbase + s * 16 + lrow) * K + kb;
      v8bf lo = *(const v8bf*)(W + base);
      v8bf hi = *(const v8bf*)(W + base + 8);
      b[s] = cat16(lo, hi);
    }
    for (int mt = 0; mt < 4; ++mt)
      for (int s = 0; s < 2; ++s)
        acc[mt][s] = __builtin_amdgcn_wmma_f32_16x16x32_bf16(
            false, a[mt], false, b[s], (short)0, acc[mt][s], false, false);
  }

  for (int mt = 0; mt < 4; ++mt)
    for (int s = 0; s < 2; ++s) {
      const int n = nbase + s * 16 + lrow;
      for (int r = 0; r < 8; ++r) {
        const int m = mbase + mt * 16 + r + 8 * half;
        out[(size_t)m * GG + n] = acc[mt][s][r];
      }
    }
}

// ---------------------------------------------------------------------------
// Persistent single-workgroup LSTM recurrence (one layer).
// 1024 threads = 32 waves on one WGP. LDS (dynamic, 256KB):
//   hbuf[2][B*H] bf16 (double buffer; read-all/write-own per step)
//   cbuf[B*H]    fp32 (each wave exclusively owns 16 hidden columns)
// Per step: gates[64,2048] = xproj_t + h @ Whh^T ; each wave computes the
// 4 gate tiles (i,f,g,o) for its 16 hidden columns across all 64 batch rows
// (two passes of 32 rows to bound register pressure), then updates c/h
// locally with no cross-wave traffic. One workgroup barrier per step.
// YMODE 0: y -> bf16 (feeds next layer GEMM). YMODE 1: y -> fp32 d_out.
// ---------------------------------------------------------------------------
template <int YMODE>
__global__ __launch_bounds__(1024)
void lstm_rec(const float* __restrict__ xproj, const bf16_t* __restrict__ Whh,
              const float* __restrict__ h0, const float* __restrict__ c0,
              bf16_t* __restrict__ ybf, float* __restrict__ yf,
              float* __restrict__ hT, float* __restrict__ cT) {
  extern __shared__ char smem[];
  bf16_t* hbuf = (bf16_t*)smem;                                   // 2*64*512 bf16
  float*  cbuf = (float*)(smem + 2 * BB * HH * sizeof(bf16_t));   // 64*512 f32

  const int tid = threadIdx.x;

  for (int i = tid; i < BB * HH; i += 1024) {
    hbuf[i] = (bf16_t)h0[i];
    cbuf[i] = c0[i];
  }
  __syncthreads();

  const int wid  = tid >> 5;
  const int lane = tid & 31;
  const int lrow = lane & 15;
  const int half = lane >> 4;
  const int ncol = wid * 16 + lrow;  // hidden column this lane owns in D tiles

  for (int t = 0; t < TSEQ; ++t) {
    bf16_t* cur = hbuf + (t & 1) * (BB * HH);
    bf16_t* nxt = hbuf + ((t + 1) & 1) * (BB * HH);
    const float* xp = xproj + (size_t)t * BB * GG;

    if (t + 1 < TSEQ)  // warm L2/GL2 for next step's pre-activations
      __builtin_prefetch(xproj + (size_t)(t + 1) * BB * GG + (size_t)tid * 128,
                         0, 1);

    for (int mp = 0; mp < 2; ++mp) {
      // accumulators: [m-tile 0..1][gate i,f,g,o] of 16x16 fp32
      v8f acc[2][4];
      for (int mt = 0; mt < 2; ++mt)
        for (int g = 0; g < 4; ++g)
          for (int r = 0; r < 8; ++r) {
            const int m = mp * 32 + mt * 16 + r + 8 * half;
            acc[mt][g][r] = xp[(size_t)m * GG + g * HH + ncol];
          }

#pragma unroll 4
      for (int kt = 0; kt < HH / 32; ++kt) {
        v16bf a[2], b[4];
        const int ka = kt * 32 + half * 16;
        for (int mt = 0; mt < 2; ++mt) {
          const int row = mp * 32 + mt * 16 + lrow;
          v8bf lo = *(const v8bf*)(cur + row * HH + ka);
          v8bf hi = *(const v8bf*)(cur + row * HH + ka + 8);
          a[mt] = cat16(lo, hi);
        }
        const int kb = kt * 32 + half * 16;
        for (int g = 0; g < 4; ++g) {
          const size_t wrow = (size_t)(g * HH + ncol) * HH + kb;
          v8bf lo = *(const v8bf*)(Whh + wrow);
          v8bf hi = *(const v8bf*)(Whh + wrow + 8);
          b[g] = cat16(lo, hi);
        }
        for (int mt = 0; mt < 2; ++mt)
          for (int g = 0; g < 4; ++g)
            acc[mt][g] = __builtin_amdgcn_wmma_f32_16x16x32_bf16(
                false, a[mt], false, b[g], (short)0, acc[mt][g], false, false);
      }

      // gate activations + cell/hidden update (wave-exclusive columns)
      for (int mt = 0; mt < 2; ++mt)
        for (int r = 0; r < 8; ++r) {
          const int m   = mp * 32 + mt * 16 + r + 8 * half;
          const int idx = m * HH + ncol;
          const float ig = sigmoidf_(acc[mt][0][r]);
          const float fg = sigmoidf_(acc[mt][1][r]);
          const float gg = tanhf(acc[mt][2][r]);
          const float og = sigmoidf_(acc[mt][3][r]);
          const float cv = fg * cbuf[idx] + ig * gg;
          cbuf[idx] = cv;
          const float hv = og * tanhf(cv);
          nxt[idx] = (bf16_t)hv;
          if (YMODE == 0)
            ybf[(size_t)t * BB * HH + idx] = (bf16_t)hv;
          else
            yf[(size_t)t * BB * HH + idx] = hv;
          if (t == TSEQ - 1) {
            hT[idx] = hv;
            cT[idx] = cv;
          }
        }
    }
    __syncthreads();  // h double-buffer flip
  }
}

// ---------------------------------------------------------------------------
// kernel_launch
// inputs: x, h0_0, c0_0, h0_1, c0_1, Wih0, Whh0, bih0, bhh0, Wih1, Whh1,
//         bih1, bhh1   (all fp32)
// d_out (fp32): y1[T*B*H] | h1f[B*H] | c1f[B*H] | h2f[B*H] | c2f[B*H]
// ---------------------------------------------------------------------------
extern "C" void kernel_launch(void* const* d_in, const int* in_sizes, int n_in,
                              void* d_out, int out_size, void* d_ws,
                              size_t ws_size, hipStream_t stream) {
  const float* x    = (const float*)d_in[0];
  const float* h0_0 = (const float*)d_in[1];
  const float* c0_0 = (const float*)d_in[2];
  const float* h0_1 = (const float*)d_in[3];
  const float* c0_1 = (const float*)d_in[4];
  const float* Wih0 = (const float*)d_in[5];
  const float* Whh0 = (const float*)d_in[6];
  const float* bih0 = (const float*)d_in[7];
  const float* bhh0 = (const float*)d_in[8];
  const float* Wih1 = (const float*)d_in[9];
  const float* Whh1 = (const float*)d_in[10];
  const float* bih1 = (const float*)d_in[11];
  const float* bhh1 = (const float*)d_in[12];
  float* out = (float*)d_out;

  char* w = (char*)d_ws;
  bf16_t* x_bf    = (bf16_t*)w;  w += (size_t)TSEQ * BB * CC * 2;
  bf16_t* Wih0_bf = (bf16_t*)w;  w += (size_t)GG * CC * 2;
  bf16_t* Whh0_bf = (bf16_t*)w;  w += (size_t)GG * HH * 2;
  bf16_t* Wih1_bf = (bf16_t*)w;  w += (size_t)GG * HH * 2;
  bf16_t* Whh1_bf = (bf16_t*)w;  w += (size_t)GG * HH * 2;
  bf16_t* y0_bf   = (bf16_t*)w;  w += (size_t)TSEQ * BB * HH * 2;
  float*  xproj   = (float*)w;   w += (size_t)TSEQ * BB * GG * 4;

  cast_f32_bf16<<<1024, 256, 0, stream>>>(x, x_bf, TSEQ * BB * CC);
  cast_f32_bf16<<<256, 256, 0, stream>>>(Wih0, Wih0_bf, GG * CC);
  cast_f32_bf16<<<256, 256, 0, stream>>>(Whh0, Whh0_bf, GG * HH);
  cast_f32_bf16<<<256, 256, 0, stream>>>(Wih1, Wih1_bf, GG * HH);
  cast_f32_bf16<<<256, 256, 0, stream>>>(Whh1, Whh1_bf, GG * HH);

  // layer 0 input projection (K = 256); LDS = 64 x K bf16 panel
  gemm_xproj<<<8192, 256, (size_t)64 * CC * 2, stream>>>(x_bf, Wih0_bf, bih0,
                                                         bhh0, xproj, CC);

  const size_t Y = (size_t)TSEQ * BB * HH;
  float* h1f = out + Y;
  float* c1f = h1f + BB * HH;
  float* h2f = c1f + BB * HH;
  float* c2f = h2f + BB * HH;

  const size_t rec_smem =
      2 * (size_t)BB * HH * sizeof(bf16_t) + (size_t)BB * HH * sizeof(float);

  // layer 0 recurrence -> y0 (bf16), final h/c fp32
  lstm_rec<0><<<1, 1024, rec_smem, stream>>>(xproj, Whh0_bf, h0_0, c0_0, y0_bf,
                                             nullptr, h1f, c1f);

  // layer 1 input projection (K = 512), reuses xproj buffer
  gemm_xproj<<<8192, 256, (size_t)64 * HH * 2, stream>>>(y0_bf, Wih1_bf, bih1,
                                                         bhh1, xproj, HH);

  // layer 1 recurrence -> y1 (fp32, straight to d_out), final h/c fp32
  lstm_rec<1><<<1, 1024, rec_smem, stream>>>(xproj, Whh1_bf, h0_1, c0_1,
                                             nullptr, out, h2f, c2f);
}
